// UnoptimizedNeuralNet_77567109366153
// MI455X (gfx1250) — matrix operand order repaired and checked by reading, not compiled, not run on hardware
//
#include <hip/hip_runtime.h>

// ---------------------------------------------------------------------------
// Fused 2-layer MLP (784 -> 256 ReLU -> 10 softmax) for MI455X / gfx1250.
// f32 reference accuracy via split-bf16 ("bf16x3") WMMA emulation:
//   a*b ~= a_hi*b_hi + a_lo*b_hi + a_hi*b_lo   (f32 accumulate)
// - Matmuls: v_wmma_f32_16x16x32_bf16, N-tiles processed in pairs so
//   consecutive WMMAs hit alternating accumulators (no RAW stalls) and
//   B-fragment ds_loads batch under a single dscnt drain.
// - X tiles: TDM tensor_load_to_lds (async, double-buffered, s_wait_tensorcnt)
// - W1 tiles: VGPR-staged, split hi/lo bf16 once on store, double-buffered
// ---------------------------------------------------------------------------

typedef __attribute__((ext_vector_type(16))) __bf16   v16bf;
typedef __attribute__((ext_vector_type(8)))  __bf16   v8bf;
typedef __attribute__((ext_vector_type(8)))  float    v8f;
typedef __attribute__((ext_vector_type(4)))  uint32_t uint32x4;
typedef __attribute__((ext_vector_type(8)))  int      int32x8;
typedef __attribute__((ext_vector_type(4)))  int      int32x4;

#define BATCH   32768
#define K_IN    784
#define HID     256
#define NCLS    10
#define M_BLK   64
#define K_TILE  64
#define N_KT    13           // ceil(784/64); last tile zero-filled by TDM OOB

// Dynamic LDS layout (bytes); 180 KB of the 320 KB/WGP.
#define XF0_OFF 0            // 64 x 64 f32   = 16384 (TDM dest, buffer 0)
#define XF1_OFF 16384        // 64 x 64 f32   = 16384 (TDM dest, buffer 1)
#define WH0_OFF 32768        // 256 x 64 bf16 = 32768 (W1 hi, buffer 0)
#define WL0_OFF 65536        // 256 x 64 bf16 = 32768 (W1 lo, buffer 0)
#define WH1_OFF 98304        // 256 x 64 bf16 = 32768 (W1 hi, buffer 1)
#define WL1_OFF 131072       // 256 x 64 bf16 = 32768 (W1 lo, buffer 1)
#define W2H_OFF 163840       // 16 x 256 bf16 = 8192 (zero-padded rows 10..15)
#define W2L_OFF 172032       // 16 x 256 bf16 = 8192
#define LG_OFF  180224       // 64 x 16 f32   = 4096
#define SMEM_BYTES 184320
// Phase 2 aliases (valid after the post-layer-1 barrier):
#define HH_OFF  0            // H hi: 64 x 256 bf16 = 32768 (over XF0+XF1)
#define HL_OFF  32768        // H lo: 64 x 256 bf16 = 32768 (over WH0)

__device__ __forceinline__ v8f zero8() {
  v8f z;
#pragma unroll
  for (int i = 0; i < 8; ++i) z[i] = 0.0f;
  return z;
}

__device__ __forceinline__ float4 ld4(const float* p, bool valid) {
  float4 z; z.x = 0.f; z.y = 0.f; z.z = 0.f; z.w = 0.f;
  return valid ? *reinterpret_cast<const float4*>(p) : z;
}

// Split 8 f32 into hi/lo bf16 halves: f ~= hi + lo, |lo| <~ 2^-9 |f|.
__device__ __forceinline__ void split8(float4 a, float4 b, v8bf& hi, v8bf& lo) {
  float f[8] = {a.x, a.y, a.z, a.w, b.x, b.y, b.z, b.w};
#pragma unroll
  for (int i = 0; i < 8; ++i) {
    __bf16 h = (__bf16)f[i];
    hi[i] = h;
    lo[i] = (__bf16)(f[i] - (float)h);
  }
}

__device__ __forceinline__ v16bf join16(v8bf lo, v8bf hi) {
  return __builtin_shufflevector(lo, hi, 0, 1, 2, 3, 4, 5, 6, 7,
                                          8, 9, 10, 11, 12, 13, 14, 15);
}

// A fragment (16x32, ISA 7.12.2 layout) split hi/lo from the f32 LDS tile.
__device__ __forceinline__ void afrag_split(const float* Xf, int row, int ks,
                                            int lhi, v16bf& ahi, v16bf& alo) {
  const float* p0 = Xf + row * K_TILE + ks + lhi * 8;
  float4 a0 = *reinterpret_cast<const float4*>(p0);
  float4 a1 = *reinterpret_cast<const float4*>(p0 + 4);
  float4 b0 = *reinterpret_cast<const float4*>(p0 + 16);
  float4 b1 = *reinterpret_cast<const float4*>(p0 + 20);
  v8bf h0, l0, h1, l1;
  split8(a0, a1, h0, l0);
  split8(b0, b1, h1, l1);
  ahi = join16(h0, h1);
  alo = join16(l0, l1);
}

// B fragment (32x16): lane n = l16, K = 8*lhi+[0,8) and 16+8*lhi+[0,8).
__device__ __forceinline__ v16bf bfrag(const __bf16* W, int n, int ks, int lhi) {
  const __bf16* bb = W + n * K_TILE + ks + lhi * 8;
  return join16(*reinterpret_cast<const v8bf*>(bb),
                *reinterpret_cast<const v8bf*>(bb + 16));
}

// TDM: async-load a 64-row x rem_k-col f32 tile (row stride 784) into LDS.
// D# per CDNA5 ISA 8.3/8.4; OOB columns beyond rem_k return zero (hardware
// zero-padding of the final K tile).
__device__ __forceinline__ void tdm_load_x(const float* gsrc, uint32_t lds_off,
                                           uint32_t rem_k) {
  const uint64_t ga = (uint64_t)(uintptr_t)gsrc;
  uint32x4 g0;
  g0[0] = 1u;                                    // count=1 (valid user D#)
  g0[1] = lds_off;                               // lds_addr (bytes)
  g0[2] = (uint32_t)ga;                          // global_addr[31:0]
  g0[3] = ((uint32_t)(ga >> 32) & 0x1FFFFFFu)    // global_addr[56:32]
          | (2u << 30);                          // type=2 ("image")
  int32x8 g1;
  g1[0] = (int)(2u << 16);                       // data_size = 4 bytes
  g1[1] = (int)((rem_k & 0xFFFFu) << 16);        // tensor_dim0[15:0]
  g1[2] = (int)(((rem_k >> 16) & 0xFFFFu)        // tensor_dim0[31:16]
                | (64u << 16));                  // tensor_dim1[15:0] = 64 rows
  g1[3] = (int)(64u << 16);                      // tile_dim0 = 64
  g1[4] = 64;                                    // tile_dim1 = 64 (tile_dim2=0)
  g1[5] = (int)K_IN;                             // tensor_dim0_stride = 784
  g1[6] = 0;
  g1[7] = 0;
  int32x4 z4; z4[0] = 0; z4[1] = 0; z4[2] = 0; z4[3] = 0;
#if defined(__clang_major__) && (__clang_major__ >= 23)
  int32x8 z8;
#pragma unroll
  for (int i = 0; i < 8; ++i) z8[i] = 0;
  __builtin_amdgcn_tensor_load_to_lds(g0, g1, z4, z4, z8, 0);
#else
  __builtin_amdgcn_tensor_load_to_lds(g0, g1, z4, z4, 0);
#endif
}

// Staged W1 tile: one 64-f32 row per thread, held in VGPRs across compute.
struct WStage { float4 v[16]; };

__device__ __forceinline__ void stage_load(const float* __restrict__ w1, int k0,
                                           WStage& s) {
  const float* src = w1 + (size_t)threadIdx.x * K_IN + k0;
#pragma unroll
  for (int g = 0; g < 16; ++g) {
    const int kg = k0 + g * 4;
    s.v[g] = ld4(src + g * 4, kg + 4 <= K_IN);
  }
}

__device__ __forceinline__ void stage_store(__bf16* Wh, __bf16* Wl,
                                            const WStage& s) {
#pragma unroll
  for (int g = 0; g < 8; ++g) {
    v8bf hi, lo;
    split8(s.v[2 * g], s.v[2 * g + 1], hi, lo);
    *reinterpret_cast<v8bf*>(&Wh[threadIdx.x * K_TILE + g * 8]) = hi;
    *reinterpret_cast<v8bf*>(&Wl[threadIdx.x * K_TILE + g * 8]) = lo;
  }
}

__global__ __launch_bounds__(256)
void mlp_fused_kernel(const float* __restrict__ x,
                      const float* __restrict__ w1,
                      const float* __restrict__ b1,
                      const float* __restrict__ w2,
                      const float* __restrict__ b2,
                      float* __restrict__ out) {
  extern __shared__ char smem[];
  __bf16* Hh  = reinterpret_cast<__bf16*>(smem + HH_OFF);
  __bf16* Hl  = reinterpret_cast<__bf16*>(smem + HL_OFF);
  __bf16* W2h = reinterpret_cast<__bf16*>(smem + W2H_OFF);
  __bf16* W2l = reinterpret_cast<__bf16*>(smem + W2L_OFF);
  float*  LG  = reinterpret_cast<float*>(smem + LG_OFF);

  const int tid   = threadIdx.x;
  const int lane  = tid & 31;         // wave32
  const int wave  = tid >> 5;         // 8 waves
  const int l16   = lane & 15;
  const int lhi   = lane >> 4;        // 0 or 1 (half-wave)
  const int mw    = wave & 3;         // 4 waves along M
  const int nw    = wave >> 2;        // 2 waves along N
  const int mbase = mw * 16;
  const int nbase = nw * 128;
  const int row0  = blockIdx.x * M_BLK;

  // ---- stage W2 split hi/lo (zero-padded 10 -> 16 rows), bf16 [16][256] ----
  {
    const int n  = tid >> 4;
    const int kk = (tid & 15) * 16;
    const bool valid = n < NCLS;
    const float* src = w2 + n * HID + kk;
    float4 a0 = ld4(src,      valid), a1 = ld4(src + 4,  valid);
    float4 b0 = ld4(src + 8,  valid), b1 = ld4(src + 12, valid);
    v8bf h0, l0, h1, l1;
    split8(a0, a1, h0, l0);
    split8(b0, b1, h1, l1);
    *reinterpret_cast<v8bf*>(&W2h[n * HID + kk])     = h0;
    *reinterpret_cast<v8bf*>(&W2h[n * HID + kk + 8]) = h1;
    *reinterpret_cast<v8bf*>(&W2l[n * HID + kk])     = l0;
    *reinterpret_cast<v8bf*>(&W2l[n * HID + kk + 8]) = l1;
  }

  // ---- prologue: fill buffer 0 (TDM for X, staged split-cvt for W1) ----
  if (wave == 0)
    tdm_load_x(x + (size_t)row0 * K_IN,
               (uint32_t)(uintptr_t)(smem + XF0_OFF), K_IN);
  {
    WStage w0s;
    stage_load(w1, 0, w0s);
    stage_store(reinterpret_cast<__bf16*>(smem + WH0_OFF),
                reinterpret_cast<__bf16*>(smem + WL0_OFF), w0s);
  }
  __builtin_amdgcn_s_wait_tensorcnt(0);
  __syncthreads();

  // ---- Layer 1: H = relu(X @ W1^T + b1), split-bf16 x3 WMMA ----
  v8f acc[8];
#pragma unroll
  for (int j = 0; j < 8; ++j) acc[j] = zero8();

  for (int kt = 0; kt < N_KT; ++kt) {
    const int  cur  = kt & 1;
    const bool more = (kt + 1 < N_KT);

    WStage wnext;
    if (more) {  // issue next tile's loads before computing current tile
      const int kn = (kt + 1) * K_TILE;
      if (wave == 0)
        tdm_load_x(x + (size_t)row0 * K_IN + kn,
                   (uint32_t)(uintptr_t)(smem + (cur ? XF0_OFF : XF1_OFF)),
                   (uint32_t)(K_IN - kn));
      stage_load(w1, kn, wnext);
    }

    const float*  Xcur = reinterpret_cast<const float*>(
        smem + (cur ? XF1_OFF : XF0_OFF));
    const __bf16* Whc = reinterpret_cast<const __bf16*>(
        smem + (cur ? WH1_OFF : WH0_OFF));
    const __bf16* Wlc = reinterpret_cast<const __bf16*>(
        smem + (cur ? WL1_OFF : WL0_OFF));

#pragma unroll
    for (int ks = 0; ks < K_TILE; ks += 32) {
      v16bf ahi, alo;
      afrag_split(Xcur, mbase + l16, ks, lhi, ahi, alo);
      // Pairs of N-tiles: batch 8 ds_loads, then 6 WMMAs alternating
      // accumulators so no consecutive WMMA has a RAW dependence.
#pragma unroll
      for (int j = 0; j < 8; j += 2) {
        const int n0 = nbase + j * 16 + l16;
        const int n1 = n0 + 16;
        v16bf bh0 = bfrag(Whc, n0, ks, lhi);
        v16bf bl0 = bfrag(Wlc, n0, ks, lhi);
        v16bf bh1 = bfrag(Whc, n1, ks, lhi);
        v16bf bl1 = bfrag(Wlc, n1, ks, lhi);
        acc[j]     = __builtin_amdgcn_wmma_f32_16x16x32_bf16(
            false, ahi, false, bh0, (short)0, acc[j],     false, false);
        acc[j + 1] = __builtin_amdgcn_wmma_f32_16x16x32_bf16(
            false, ahi, false, bh1, (short)0, acc[j + 1], false, false);
        acc[j]     = __builtin_amdgcn_wmma_f32_16x16x32_bf16(
            false, alo, false, bh0, (short)0, acc[j],     false, false);
        acc[j + 1] = __builtin_amdgcn_wmma_f32_16x16x32_bf16(
            false, alo, false, bh1, (short)0, acc[j + 1], false, false);
        acc[j]     = __builtin_amdgcn_wmma_f32_16x16x32_bf16(
            false, ahi, false, bl0, (short)0, acc[j],     false, false);
        acc[j + 1] = __builtin_amdgcn_wmma_f32_16x16x32_bf16(
            false, ahi, false, bl1, (short)0, acc[j + 1], false, false);
      }
    }

    if (more)
      stage_store(reinterpret_cast<__bf16*>(smem + (cur ? WH0_OFF : WH1_OFF)),
                  reinterpret_cast<__bf16*>(smem + (cur ? WL0_OFF : WL1_OFF)),
                  wnext);
    __builtin_amdgcn_s_wait_tensorcnt(0);  // fence wave0's TDM before barrier
    __syncthreads();
  }

  // bias + ReLU, store H split hi/lo bf16 [64][256] (aliases dead buffers)
#pragma unroll
  for (int j = 0; j < 8; ++j) {
    const int n = nbase + j * 16 + l16;
    const float bias = b1[n];
#pragma unroll
    for (int r = 0; r < 8; ++r) {
      const int m = mbase + r + lhi * 8;  // C layout: VGPR r -> rows r / r+8
      float h = acc[j][r] + bias;
      h = h > 0.0f ? h : 0.0f;
      __bf16 hb = (__bf16)h;
      Hh[m * HID + n] = hb;
      Hl[m * HID + n] = (__bf16)(h - (float)hb);
    }
  }
  __syncthreads();

  // ---- Layer 2: logits = H @ W2^T (N padded to 16), waves 0..3 ----
  // Two partial accumulators (even/odd K-steps) keep WMMAs independent.
  if (wave < 4) {
    const int mb2 = wave * 16;
    v8f lacc0 = zero8();
    v8f lacc1 = zero8();
#pragma unroll
    for (int s = 0; s < 8; s += 2) {
#pragma unroll
      for (int t = 0; t < 2; ++t) {
        const int ks = (s + t) * 32;
        const __bf16* ph = &Hh[(mb2 + l16) * HID + ks + lhi * 8];
        const __bf16* pl = &Hl[(mb2 + l16) * HID + ks + lhi * 8];
        v16bf ahi = join16(*reinterpret_cast<const v8bf*>(ph),
                           *reinterpret_cast<const v8bf*>(ph + 16));
        v16bf alo = join16(*reinterpret_cast<const v8bf*>(pl),
                           *reinterpret_cast<const v8bf*>(pl + 16));
        const __bf16* qh = &W2h[l16 * HID + ks + lhi * 8];
        const __bf16* ql = &W2l[l16 * HID + ks + lhi * 8];
        v16bf bhi = join16(*reinterpret_cast<const v8bf*>(qh),
                           *reinterpret_cast<const v8bf*>(qh + 16));
        v16bf blo = join16(*reinterpret_cast<const v8bf*>(ql),
                           *reinterpret_cast<const v8bf*>(ql + 16));
        v8f& lac = t ? lacc1 : lacc0;
        lac = __builtin_amdgcn_wmma_f32_16x16x32_bf16(
            false, ahi, false, bhi, (short)0, lac, false, false);
        lac = __builtin_amdgcn_wmma_f32_16x16x32_bf16(
            false, alo, false, bhi, (short)0, lac, false, false);
        lac = __builtin_amdgcn_wmma_f32_16x16x32_bf16(
            false, ahi, false, blo, (short)0, lac, false, false);
      }
    }
#pragma unroll
    for (int r = 0; r < 8; ++r) {
      const int m = mb2 + r + lhi * 8;
      LG[m * 16 + l16] = lacc0[r] + lacc1[r];
    }
  }
  __syncthreads();

  // ---- softmax over the 10 real classes, one row per thread ----
  if (tid < M_BLK) {
    const int m = tid;
    const size_t grow = (size_t)row0 + m;
    float v[NCLS];
    float mx = -3.4e38f;
#pragma unroll
    for (int c = 0; c < NCLS; ++c) {
      v[c] = LG[m * 16 + c] + b2[c];
      mx = fmaxf(mx, v[c]);
    }
    float s = 0.0f;
#pragma unroll
    for (int c = 0; c < NCLS; ++c) {
      v[c] = __expf(v[c] - mx);
      s += v[c];
    }
    const float inv = 1.0f / s;
#pragma unroll
    for (int c = 0; c < NCLS; ++c) out[grow * NCLS + c] = v[c] * inv;
  }
}

extern "C" void kernel_launch(void* const* d_in, const int* in_sizes, int n_in,
                              void* d_out, int out_size, void* d_ws, size_t ws_size,
                              hipStream_t stream) {
  (void)in_sizes; (void)n_in; (void)out_size; (void)d_ws; (void)ws_size;
  const float* x  = (const float*)d_in[0];
  const float* w1 = (const float*)d_in[1];
  const float* b1 = (const float*)d_in[2];
  const float* w2 = (const float*)d_in[3];
  const float* b2 = (const float*)d_in[4];
  float* out = (float*)d_out;

  dim3 grid(BATCH / M_BLK);  // 512 workgroups
  dim3 block(256);           // 8 wave32 waves
  mlp_fused_kernel<<<grid, block, SMEM_BYTES, stream>>>(x, w1, b1, w2, b2, out);
}
